// GCNThreeOmics_40407052320948
// MI455X (gfx1250) — matrix operand order repaired and checked by reading, not compiled, not run on hardware
//
#include <hip/hip_runtime.h>
#include <hip/hip_bf16.h>
#include <math.h>
#include <stdint.h>
#include <type_traits>

// ---------------------------------------------------------------------------
// CDNA5 / gfx1250 bf16 WMMA GEMM pipeline for the 3-omics GCN.
//   - all matmuls: v_wmma_f32_16x16x32_bf16, fp32 accumulate
//   - adj (1 GiB fp32) streamed exactly once per graph-conv (tile spans full N)
//   - A tiles: register-staged (fp32 -> bf16 convert), double-barrier publish
//   - B tiles: Tensor Data Mover (tensor_load_to_lds) into double-buffered,
//     hw-padded LDS; completion via TENSORcnt
// ---------------------------------------------------------------------------

typedef __attribute__((ext_vector_type(16))) __bf16 v16bf;
typedef __attribute__((ext_vector_type(8)))  __bf16 v8bf;
typedef __attribute__((ext_vector_type(4)))  __bf16 v4bf;
typedef __attribute__((ext_vector_type(8)))  float  v8f;
typedef __attribute__((ext_vector_type(4)))  unsigned int v4u;
typedef __attribute__((ext_vector_type(8)))  unsigned int v8u;

#define LSTR 40   // LDS row pitch in bf16 elements (64B data + 16B pad = 80B)

// Build a 16-element bf16 WMMA fragment from two 16B LDS reads.
__device__ __forceinline__ v16bf ld_frag16(const __bf16* p0, const __bf16* p1) {
  v8bf lo = *(const v8bf*)p0;
  v8bf hi = *(const v8bf*)p1;
  return __builtin_shufflevector(lo, hi, 0,1,2,3,4,5,6,7,8,9,10,11,12,13,14,15);
}

// TDM: DMA a [tileN rows x 32 cols] bf16 tile of tensor [Nrows][K] (row-major)
// into LDS at ldsAddr with 16B padding after every 64B row (pitch = 80B).
// OOB columns (k >= K) are written as zero by hardware.
// D# bitfields per CDNA5 ISA ch.8; groups 2/3 unused (2-D tensor) -> zeros.
__device__ __forceinline__ void tdm_load_b(unsigned ldsAddr,
                                           const __bf16* gaddr,
                                           int K, int Nrows, int tileN) {
  const unsigned long long ga = (unsigned long long)(uintptr_t)gaddr;
  v4u g0;
  g0[0] = 1u;                                   // count=1 (valid), user mode
  g0[1] = ldsAddr;                              // lds_addr [63:32]
  g0[2] = (unsigned)(ga & 0xFFFFFFFFu);         // global_addr[31:0]
  g0[3] = (unsigned)((ga >> 32) & 0x01FFFFFFu)  // global_addr[56:32]
        | (2u << 30);                           // type = 2 ("image")
  v8u g1;
  g1[0] = (1u << 16)                            // data_size = 1 (2 bytes)
        | (1u << 20)                            // pad_enable
        | (3u << 22)                            // pad_interval: 16 DW (64B)
        | (3u << 25);                           // pad_amount: 4 DW (16B)
  g1[1] = ((unsigned)K & 0xFFFFu) << 16;        // tensor_dim0[15:0]
  g1[2] = (((unsigned)K >> 16) & 0xFFFFu)       // tensor_dim0[31:16]
        | (((unsigned)Nrows & 0xFFFFu) << 16);  // tensor_dim1[15:0]
  g1[3] = (((unsigned)Nrows >> 16) & 0xFFFFu)   // tensor_dim1[31:16]
        | (32u << 16);                          // tile_dim0 = 32 elements
  g1[4] = (unsigned)tileN & 0xFFFFu;            // tile_dim1 ; tile_dim2 = 0
  g1[5] = (unsigned)K;                          // tensor_dim0_stride[31:0]
  g1[6] = 0u;                                   // stride0[47:32], stride1 lo
  g1[7] = 0u;
  const v4u gz = {0u, 0u, 0u, 0u};              // groups 2/3: unused (2-D)
  asm volatile("tensor_load_to_lds %0, %1, %2, %3"
               :: "s"(g0), "s"(g1), "s"(gz), "s"(gz)
               : "memory");
}

// Pre-transpose + fp32->bf16 convert:  in [K][N] row-major -> out [N][K].
__global__ __launch_bounds__(256)
void k_transpose_cvt(const float* __restrict__ in, __bf16* __restrict__ out,
                     int K, int Nc) {
  const int idx = blockIdx.x * 256 + threadIdx.x;
  if (idx >= K * Nc) return;
  const int k = idx / Nc;
  const int n = idx - k * Nc;
  out[(size_t)n * (size_t)K + (size_t)k] = (__bf16)in[idx];
}

// Row-wise log_softmax for [rows][32]; one wave32 per row (lane == class).
__global__ __launch_bounds__(256)
void k_logsoftmax32(const float* __restrict__ in, float* __restrict__ out) {
  const int row  = blockIdx.x * 8 + (threadIdx.x >> 5);
  const int lane = threadIdx.x & 31;
  const float v = in[row * 32 + lane];
  float m = v;
  #pragma unroll
  for (int d = 16; d > 0; d >>= 1) m = fmaxf(m, __shfl_xor(m, d, 32));
  const float e = __expf(v - m);
  float s = e;
  #pragma unroll
  for (int d = 16; d > 0; d >>= 1) s += __shfl_xor(s, d, 32);
  out[row * 32 + lane] = (v - m) - __logf(s);
}

// Tiled WMMA GEMM:  C = epilogue(A[M,K] * Bt^T + bias)
//   A  : fp32 (converted to bf16 on stage-in) or bf16, row-major, ld = lda
//   Bt : bf16, PRE-TRANSPOSED [N][K] row-major (B-fragments = contiguous K)
//   C  : bf16 row-major (ldc) | bf16 transposed [N][M] | fp32 row-major
// Block: WM*WN wave32 waves; wave tile 32x32 (2x2 WMMA frags).
// Block tile = (WM*32) x (WN*32).  EVEN_K removes A-side K-tail guards
// (B-side tail handled for free by TDM OOB-zero semantics).
template<int WM, int WN, bool A_F32, bool EVEN_K, bool HAS_BIAS, bool RELU,
         bool STORE_T, bool OUT_F32>
__global__ __launch_bounds__(WM * WN * 32)
void gemm_wmma_bf16(const void* __restrict__ Aptr, int lda,
                    const __bf16* __restrict__ Bt,
                    const float* __restrict__ bias,
                    void* __restrict__ Cptr, int ldc,
                    int M, int N, int K)
{
  constexpr int NTHR = WM * WN * 32;
  constexpr int BM = WM * 32;
  constexpr int BN = WN * 32;
  constexpr int AQ = (BM * 8) / NTHR;   // 4-elem staging slots per thread (A)
  __shared__ __bf16 lsA[BM * LSTR];
  __shared__ __bf16 lsB[2][BN * LSTR];  // double-buffered, TDM-filled

  const int tid  = threadIdx.x;
  const int lane = tid & 31;
  const int l15  = lane & 15;
  const bool hiHalf = lane >= 16;
  const int w  = tid >> 5;
  const int m0 = (w / WN) * 32;
  const int n0 = (w % WN) * 32;
  const int blockM = blockIdx.x * BM;
  const int blockN = blockIdx.y * BN;

  using AElem = typename std::conditional<A_F32, float, __bf16>::type;
  using AReg  = typename std::conditional<A_F32, float4, v4bf>::type;

  // ---- hoisted per-thread A staging addresses ---------------------------
  int ldsOffA[AQ], kqA[AQ];
  const AElem* rowA[AQ];
  #pragma unroll
  for (int q = 0; q < AQ; ++q) {
    const int i = tid + q * NTHR;
    const int r = i >> 3;
    kqA[q] = (i & 7) << 2;
    ldsOffA[q] = r * LSTR + kqA[q];
    rowA[q] = (const AElem*)Aptr + (size_t)(blockM + r) * (size_t)lda;
  }

  AReg regA[AQ];

  auto loadRegsA = [&](int k0) {
    #pragma unroll
    for (int q = 0; q < AQ; ++q) {
      const int gk = k0 + kqA[q];
      if constexpr (A_F32) {
        float4 f;
        if (EVEN_K || gk + 3 < K) {
          f = *(const float4*)(rowA[q] + gk);
        } else {
          f.x = (gk + 0 < K) ? rowA[q][gk + 0] : 0.f;
          f.y = (gk + 1 < K) ? rowA[q][gk + 1] : 0.f;
          f.z = (gk + 2 < K) ? rowA[q][gk + 2] : 0.f;
          f.w = (gk + 3 < K) ? rowA[q][gk + 3] : 0.f;
        }
        regA[q] = f;
      } else {
        v4bf bv;
        if (EVEN_K || gk + 3 < K) {
          bv = *(const v4bf*)(rowA[q] + gk);
        } else {
          #pragma unroll
          for (int j = 0; j < 4; ++j)
            bv[j] = (gk + j < K) ? rowA[q][gk + j] : (__bf16)0.f;
        }
        regA[q] = bv;
      }
    }
  };

  auto storeLdsA = [&]() {
    #pragma unroll
    for (int q = 0; q < AQ; ++q) {
      if constexpr (A_F32) {
        const float4 f = regA[q];
        v4bf bv;
        bv[0] = (__bf16)f.x; bv[1] = (__bf16)f.y;
        bv[2] = (__bf16)f.z; bv[3] = (__bf16)f.w;
        *(v4bf*)&lsA[ldsOffA[q]] = bv;
      } else {
        *(v4bf*)&lsA[ldsOffA[q]] = regA[q];
      }
    }
  };

  // TDM source/dest bases for the B panel
  const __bf16* Bbase = Bt + (size_t)blockN * (size_t)K;
  const unsigned ldsB0 = (unsigned)(uintptr_t)&lsB[0][0];
  const unsigned ldsB1 = (unsigned)(uintptr_t)&lsB[1][0];

  const v8f vz = {0.f, 0.f, 0.f, 0.f, 0.f, 0.f, 0.f, 0.f};
  v8f acc[2][2];
  acc[0][0] = vz; acc[0][1] = vz; acc[1][0] = vz; acc[1][1] = vz;

  // hoisted fragment LDS offsets
  const int kbA = hiHalf ? 16 : 0;
  const int kbB = hiHalf ? 16 : 0;
  int fA[2], fB[2];
  #pragma unroll
  for (int ti = 0; ti < 2; ++ti) fA[ti] = (m0 + ti * 16 + l15) * LSTR + kbA;
  #pragma unroll
  for (int tj = 0; tj < 2; ++tj) fB[tj] = (n0 + tj * 16 + l15) * LSTR + kbB;

  const int KT = (K + 31) >> 5;

  // ---- prologue: TDM B tile 0, reg-stage A tile 0 -----------------------
  if (w == 0) tdm_load_b(ldsB0, Bbase, K, N, BN);
  loadRegsA(0);
  storeLdsA();
  __builtin_amdgcn_s_wait_tensorcnt(0);
  __syncthreads();

  for (int kt = 0; kt < KT; ++kt) {
    const bool more = (kt + 1 < KT);
    if (more) {
      loadRegsA((kt + 1) << 5);            // global loads overlap with WMMAs
      if (w == 0)                           // DMA next B tile into other buf
        tdm_load_b((kt + 1) & 1 ? ldsB1 : ldsB0,
                   Bbase + ((size_t)(kt + 1) << 5), K, N, BN);
    }

    // A 16x32: lanes 0-15 row M=l, K 0..15; lanes 16-31 K 16..31
    // B 32x16: lanes 0-15 col N=l, K 0..15; lanes 16-31 K 16..31
    const __bf16* lb = &lsB[kt & 1][0];
    v16bf af[2], bfr[2];
    #pragma unroll
    for (int ti = 0; ti < 2; ++ti)
      af[ti] = ld_frag16(&lsA[fA[ti]], &lsA[fA[ti] + 8]);
    #pragma unroll
    for (int tj = 0; tj < 2; ++tj)
      bfr[tj] = ld_frag16(&lb[fB[tj]], &lb[fB[tj] + 8]);

    #pragma unroll
    for (int ti = 0; ti < 2; ++ti)
      #pragma unroll
      for (int tj = 0; tj < 2; ++tj)
        acc[ti][tj] = __builtin_amdgcn_wmma_f32_16x16x32_bf16(
            false, af[ti], false, bfr[tj], (short)0, acc[ti][tj], false, false);

    if (more) {
      __syncthreads();                      // all done reading lsA tile kt
      storeLdsA();                          // publish A tile kt+1
      __builtin_amdgcn_s_wait_tensorcnt(0); // B tile kt+1 landed (wave 0)
      __syncthreads();                      // publish to all waves
    }
  }

  // ---- epilogue ----------------------------------------------------------
  // C/D layout: element r -> M = r (+8 for lanes 16-31), N = lane&15.
  #pragma unroll
  for (int ti = 0; ti < 2; ++ti) {
    #pragma unroll
    for (int tj = 0; tj < 2; ++tj) {
      const int col = blockN + n0 + tj * 16 + l15;
      float bv = 0.f;
      if constexpr (HAS_BIAS) bv = bias[col];
      const int rowBase = blockM + m0 + ti * 16 + (hiHalf ? 8 : 0);
      if constexpr (STORE_T) {
        // consecutive r -> consecutive M rows -> contiguous in [N][M]: 16B store
        v8bf ov;
        #pragma unroll
        for (int r = 0; r < 8; ++r) {
          float v = acc[ti][tj][r] + bv;
          if constexpr (RELU) v = fmaxf(v, 0.f);
          ov[r] = (__bf16)v;
        }
        *(v8bf*)&((__bf16*)Cptr)[(size_t)col * (size_t)M + rowBase] = ov;
      } else {
        #pragma unroll
        for (int r = 0; r < 8; ++r) {
          const int row = rowBase + r;
          float v = acc[ti][tj][r] + bv;
          if constexpr (RELU) v = fmaxf(v, 0.f);
          if constexpr (OUT_F32)
            ((float*)Cptr)[(size_t)row * (size_t)ldc + col] = v;
          else
            ((__bf16*)Cptr)[(size_t)row * (size_t)ldc + col] = (__bf16)v;
        }
      }
    }
  }
}

// ---------------------------------------------------------------------------

extern "C" void kernel_launch(void* const* d_in, const int* in_sizes, int n_in,
                              void* d_out, int out_size, void* d_ws, size_t ws_size,
                              hipStream_t stream) {
  (void)in_sizes; (void)n_in; (void)out_size; (void)ws_size;
  constexpr int NN = 16384;
  constexpr int F1 = 2000, F2 = 2000, F3 = 1000;
  constexpr int H = 256, HSUM = 768, NHID = 256, NCLASS = 32;

  const float* x   = (const float*)d_in[0];
  const float* adj = (const float*)d_in[1];
  const float* W1  = (const float*)d_in[2];
  const float* b1  = (const float*)d_in[3];
  const float* W2  = (const float*)d_in[4];
  const float* b2  = (const float*)d_in[5];
  const float* W3  = (const float*)d_in[6];
  const float* b3  = (const float*)d_in[7];
  const float* gw1 = (const float*)d_in[8];
  const float* gb1 = (const float*)d_in[9];
  const float* gw2 = (const float*)d_in[10];
  const float* gb2 = (const float*)d_in[11];

  char* wsb = (char*)d_ws;
  size_t off = 0;
  auto carve = [&](size_t bytes) -> void* {
    void* p = wsb + off;
    off = (off + bytes + 255) & ~(size_t)255;
    return p;
  };
  __bf16* W1t  = (__bf16*)carve((size_t)F1 * H * 2);        // [H][F1]
  __bf16* W2t  = (__bf16*)carve((size_t)F2 * H * 2);        // [H][F2]
  __bf16* W3t  = (__bf16*)carve((size_t)F3 * H * 2);        // [H][F3]
  __bf16* gw1t = (__bf16*)carve((size_t)HSUM * NHID * 2);   // [NHID][HSUM]
  __bf16* gw2t = (__bf16*)carve((size_t)NHID * NCLASS * 2); // [NCLASS][NHID]
  __bf16* h    = (__bf16*)carve((size_t)NN * HSUM * 2);     // [N][768] row-major
  __bf16* hg1T = (__bf16*)carve((size_t)NHID * NN * 2);     // [NHID][N]
  __bf16* h2   = (__bf16*)carve((size_t)NN * NHID * 2);     // [N][NHID] row-major
  __bf16* hg2T = (__bf16*)carve((size_t)NCLASS * NN * 2);   // [NCLASS][N]
  float*  lg   = (float*) carve((size_t)NN * NCLASS * 4);   // logits fp32

  // weight transpose+convert (tiny)
  auto tc = [&](const float* in, __bf16* out, int K, int Nc) {
    const int e = K * Nc;
    k_transpose_cvt<<<dim3((e + 255) / 256), dim3(256), 0, stream>>>(in, out, K, Nc);
  };
  tc(W1, W1t, F1, H);
  tc(W2, W2t, F2, H);
  tc(W3, W3t, F3, H);
  tc(gw1, gw1t, HSUM, NHID);
  tc(gw2, gw2t, NHID, NCLASS);

  // Wide config: 16 waves (512 thr), block tile 64 x 256 (full N) -> A is
  // streamed exactly once per GEMM (adj: 1 GiB single pass; B panel in L2).
  dim3 gWide(NN / 64, 1);
  // omics branches: h[:, c:c+256] = relu(x_slice @ W + b)  (K = 2000/1000)
  gemm_wmma_bf16<2, 8, true,  false, true,  true,  false, false>
      <<<gWide, dim3(512), 0, stream>>>(x,           5000, W1t, b1, (void*)(h),         HSUM, NN, H, F1);
  gemm_wmma_bf16<2, 8, true,  false, true,  true,  false, false>
      <<<gWide, dim3(512), 0, stream>>>(x + F1,      5000, W2t, b2, (void*)(h + H),     HSUM, NN, H, F2);
  gemm_wmma_bf16<2, 8, true,  false, true,  true,  false, false>
      <<<gWide, dim3(512), 0, stream>>>(x + F1 + F2, 5000, W3t, b3, (void*)(h + 2 * H), HSUM, NN, H, F3);
  // hg1 = h @ gw1  (K=768 even; transposed store -> B operand of adj GEMM)
  gemm_wmma_bf16<2, 8, false, true,  false, false, true,  false>
      <<<gWide, dim3(512), 0, stream>>>(h, HSUM, gw1t, nullptr, (void*)hg1T, 0, NN, NHID, HSUM);
  // h2 = relu(adj @ hg1 + gb1)   (adj single pass, K=16384 even)
  gemm_wmma_bf16<2, 8, true,  true,  true,  true,  false, false>
      <<<gWide, dim3(512), 0, stream>>>(adj, NN, hg1T, gb1, (void*)h2, NHID, NN, NHID, NN);

  // hg2 = h2 @ gw2  (K=256 even; 8x1 waves, tile 256x32)
  gemm_wmma_bf16<8, 1, false, true,  false, false, true,  false>
      <<<dim3(NN / 256, 1), dim3(256), 0, stream>>>(h2, NHID, gw2t, nullptr, (void*)hg2T, 0, NN, NCLASS, NHID);
  // logits = adj @ hg2 + gb2  (adj single pass; 4x1 waves, tile 128x32)
  gemm_wmma_bf16<4, 1, true,  true,  true,  false, false, true>
      <<<dim3(NN / 128, 1), dim3(128), 0, stream>>>(adj, NN, hg2T, gb2, (void*)lg, NCLASS, NN, NCLASS, NN);

  // log_softmax rows -> d_out
  k_logsoftmax32<<<dim3(NN / 8), dim3(256), 0, stream>>>(lg, (float*)d_out);
}
